// MultiHeadAttention_72584947303107
// MI455X (gfx1250) — compile-verified, matrix operand order checked by reading, not run on hardware
//
#include <hip/hip_runtime.h>

// ---------------------------------------------------------------------------
// MHA forward for B=4, S=2048, D=1024, H=16, DK=64 on gfx1250 (wave32, WMMA).
// All matmuls via v_wmma_f32_16x16x32_bf16 (fp32->bf16 inputs, fp32 accum).
// ---------------------------------------------------------------------------

typedef __attribute__((ext_vector_type(16))) __bf16 bf16x16;
typedef __attribute__((ext_vector_type(8)))  __bf16 bf16x8;
typedef __attribute__((ext_vector_type(8)))  float  f32x8;

union Frag {
  bf16x16 v;
  bf16x8  h[2];
};

#define WMMA_BF16(a, b, c) \
  __builtin_amdgcn_wmma_f32_16x16x32_bf16(false, (a), false, (b), (short)0, (c), false, false)

static constexpr int kB  = 4;
static constexpr int kS  = 2048;
static constexpr int kD  = 1024;
static constexpr int kH  = 16;
static constexpr int kDK = 64;
static constexpr int kM  = kB * kS;   // 8192 rows for the projection GEMMs

// ---------------------------------------------------------------------------
// f32 -> bf16 elementwise convert
// ---------------------------------------------------------------------------
__global__ __launch_bounds__(256) void cvt_bf16_kernel(const float* __restrict__ src,
                                                       __bf16* __restrict__ dst, int n) {
  int i = blockIdx.x * blockDim.x + threadIdx.x;
  int stride = gridDim.x * blockDim.x;
  for (; i < n; i += stride) dst[i] = (__bf16)src[i];
}

// ---------------------------------------------------------------------------
// W [K][N] f32 row-major  ->  WT [N][K] bf16 row-major (so B fragments load
// K-contiguous, exactly like A fragments).
// ---------------------------------------------------------------------------
__global__ __launch_bounds__(256) void transpose_cvt_kernel(const float* __restrict__ W,
                                                            __bf16* __restrict__ WT) {
  int i = blockIdx.x * 256 + threadIdx.x;   // 0 .. D*D-1
  int k = i >> 10;                          // row of W   (contraction dim)
  int n = i & 1023;                         // col of W   (output dim)
  WT[(size_t)n * kD + k] = (__bf16)W[i];
}

// ---------------------------------------------------------------------------
// GEMM: out[M=8192][N=1024] = A[M][K=1024] (bf16) x W  (+ bias)
// W given transposed as BT[N][K] bf16.  out is bf16 (intermediates) or f32.
// Block = 256 thr = 8 waves; tile 128x128; K-step 32.
// Wave w: M offset (w&1)*64, N offset (w>>1)*32 -> 4x2 grid of 16x16 WMMAs.
// ---------------------------------------------------------------------------
__global__ __launch_bounds__(256) void gemm_bf16_kernel(const __bf16* __restrict__ A,
                                                        const __bf16* __restrict__ BT,
                                                        const float*  __restrict__ bias,
                                                        float*  __restrict__ outF,
                                                        __bf16* __restrict__ outB) {
  constexpr int K = kD, N = kD, TM = 128, TN = 128, TK = 32;
  __shared__ alignas(16) __bf16 As[TM][TK];
  __shared__ alignas(16) __bf16 Bs[TN][TK];

  const int tid  = threadIdx.x;
  const int lane = tid & 31;
  const int w    = tid >> 5;
  const int l15  = lane & 15;
  const int hi4  = lane >> 4;           // 0 for lanes 0-15, 1 for lanes 16-31
  const int m0   = blockIdx.y * TM;
  const int n0   = blockIdx.x * TN;
  const int wm   = (w & 1) * 64;
  const int wn   = (w >> 1) * 32;

  f32x8 acc[4][2];
#pragma unroll
  for (int i = 0; i < 4; ++i)
#pragma unroll
    for (int j = 0; j < 2; ++j)
#pragma unroll
      for (int r = 0; r < 8; ++r) acc[i][j][r] = 0.0f;

  for (int k0 = 0; k0 < K; k0 += TK) {
    __syncthreads();
    // Stage tiles: 512 chunks of 16B each for A and BT -> 2 chunks/thread each.
#pragma unroll
    for (int i = 0; i < 2; ++i) {
      int c  = tid + i * 256;           // 0..511
      int r  = c >> 2;                  // tile row 0..127
      int cc = c & 3;                   // 8-element chunk in row
      *(bf16x8*)&As[r][cc * 8] = *(const bf16x8*)&A [(size_t)(m0 + r) * K + k0 + cc * 8];
      *(bf16x8*)&Bs[r][cc * 8] = *(const bf16x8*)&BT[(size_t)(n0 + r) * K + k0 + cc * 8];
    }
    if (k0 + TK < K) {
      __builtin_prefetch(&A [(size_t)(m0 + (tid >> 2)) * K + k0 + TK], 0, 0);
      __builtin_prefetch(&BT[(size_t)(n0 + (tid >> 2)) * K + k0 + TK], 0, 0);
    }
    __syncthreads();

    Frag af[4], bf[2];
#pragma unroll
    for (int i = 0; i < 4; ++i) {
      const bf16x8* row = (const bf16x8*)&As[wm + i * 16 + l15][0];
      af[i].h[0] = row[hi4];            // K = (hi4? 8:0)  .. +7
      af[i].h[1] = row[2 + hi4];        // K = 16 + (hi4? 8:0) .. +7
    }
#pragma unroll
    for (int j = 0; j < 2; ++j) {
      const bf16x8* row = (const bf16x8*)&Bs[wn + j * 16 + l15][0];
      bf[j].h[0] = row[hi4];
      bf[j].h[1] = row[2 + hi4];
    }
#pragma unroll
    for (int i = 0; i < 4; ++i)
#pragma unroll
      for (int j = 0; j < 2; ++j)
        acc[i][j] = WMMA_BF16(af[i].v, bf[j].v, acc[i][j]);
  }

  // Epilogue: C/D layout -> lanes 0-15 hold M=r, lanes 16-31 hold M=8+r, N=lane&15.
#pragma unroll
  for (int i = 0; i < 4; ++i)
#pragma unroll
    for (int j = 0; j < 2; ++j) {
      int nn = n0 + wn + j * 16 + l15;
      float bv = bias[nn];
#pragma unroll
      for (int r = 0; r < 8; ++r) {
        int mm  = m0 + wm + i * 16 + r + hi4 * 8;
        float v = acc[i][j][r] + bv;
        if (outF) outF[(size_t)mm * N + nn] = v;
        else      outB[(size_t)mm * N + nn] = (__bf16)v;
      }
    }
}

// ---------------------------------------------------------------------------
// Flash attention: one block per (b, h, 128-query tile); 8 waves, each wave
// owns 16 query rows.  Q/K/V are bf16 [B,S,D] (head h at column h*64).
// ---------------------------------------------------------------------------
__global__ __launch_bounds__(256) void attention_kernel(const __bf16* __restrict__ Q,
                                                        const __bf16* __restrict__ Kg,
                                                        const __bf16* __restrict__ V,
                                                        __bf16* __restrict__ O) {
  // K-tile and per-wave P buffers share storage (K-tile dead once P is built).
  __shared__ alignas(16) union {
    __bf16 kt[128][64];       // 16 KB : key tile, row-major [key][d]
    __bf16 p[8][16][128];     // 32 KB : per-wave P (16 q x 128 keys)
  } sh;
  __shared__ alignas(16) __bf16 vt[64][128];   // 16 KB : V tile transposed [d][key]

  const int tid  = threadIdx.x;
  const int lane = tid & 31;
  const int w    = tid >> 5;
  const int l15  = lane & 15;
  const int hi4  = lane >> 4;
  const int qt   = blockIdx.x;          // query tile (0..15)
  const int h    = blockIdx.y;
  const int b    = blockIdx.z;
  const int q0   = qt * 128;
  const size_t base = (size_t)b * kS * kD + (size_t)h * kDK;  // + s*D per row

  // Load this wave's Q fragments (16 rows x 64 d) once, straight from global.
  Frag qf[2];
  {
    const bf16x8* row = (const bf16x8*)&Q[base + (size_t)(q0 + w * 16 + l15) * kD];
#pragma unroll
    for (int ks = 0; ks < 2; ++ks) {
      qf[ks].h[0] = row[ks * 4 + hi4];
      qf[ks].h[1] = row[ks * 4 + 2 + hi4];
    }
  }

  f32x8 o[4];
  float mrow[8], lrow[8];
#pragma unroll
  for (int d = 0; d < 4; ++d)
#pragma unroll
    for (int r = 0; r < 8; ++r) o[d][r] = 0.0f;
#pragma unroll
  for (int r = 0; r < 8; ++r) { mrow[r] = -1e30f; lrow[r] = 0.0f; }

  const float scale = 0.125f;           // 1/sqrt(64)

  for (int j = 0; j <= qt; ++j) {       // causal: only key tiles <= query tile
    const int kb = j * 128;
    __syncthreads();                    // prior iteration done with sh.p / vt
    // Stage K tile (row-major) and V tile (transposed) : 1024 16B chunks.
#pragma unroll
    for (int i = 0; i < 4; ++i) {
      int c  = tid + i * 256;           // 0..1023
      int r  = c >> 3;                  // key row 0..127
      int cc = c & 7;                   // d chunk
      *(bf16x8*)&sh.kt[r][cc * 8] = *(const bf16x8*)&Kg[base + (size_t)(kb + r) * kD + cc * 8];
      bf16x8 vv = *(const bf16x8*)&V[base + (size_t)(kb + r) * kD + cc * 8];
#pragma unroll
      for (int e = 0; e < 8; ++e) vt[cc * 8 + e][r] = vv[e];
    }
    __syncthreads();

    // S = Q K^T : 2 k-steps (d) x 8 key subtiles.  B-fragment for K^T reads
    // K rows (key = WMMA column) K-contiguous, same chunk pattern as A.
    f32x8 s[8];
#pragma unroll
    for (int n = 0; n < 8; ++n)
#pragma unroll
      for (int r = 0; r < 8; ++r) s[n][r] = 0.0f;
#pragma unroll
    for (int ks = 0; ks < 2; ++ks)
#pragma unroll
      for (int n = 0; n < 8; ++n) {
        Frag bfr;
        const bf16x8* row = (const bf16x8*)&sh.kt[n * 16 + l15][0];
        bfr.h[0] = row[ks * 4 + hi4];
        bfr.h[1] = row[ks * 4 + 2 + hi4];
        s[n] = WMMA_BF16(qf[ks].v, bfr.v, s[n]);
      }

    // Scale + causal mask (-1e9, matching reference).
    const int qi = q0 + w * 16 + hi4 * 8;     // + r gives query index
#pragma unroll
    for (int n = 0; n < 8; ++n) {
      int ki = kb + n * 16 + l15;
#pragma unroll
      for (int r = 0; r < 8; ++r) {
        float v = s[n][r] * scale;
        if (ki > qi + r) v -= 1e9f;
        s[n][r] = v;
      }
    }

    // Online softmax.  Row m = r + 8*hi4 lives across the 16 lanes of a
    // half-wave; reduce with xor over lane bits 0..3 (wave32).
    float corr[8];
#pragma unroll
    for (int r = 0; r < 8; ++r) {
      float mx = s[0][r];
#pragma unroll
      for (int n = 1; n < 8; ++n) mx = fmaxf(mx, s[n][r]);
#pragma unroll
      for (int d = 1; d < 16; d <<= 1) mx = fmaxf(mx, __shfl_xor(mx, d, 32));
      float mnew = fmaxf(mrow[r], mx);
      corr[r] = __expf(mrow[r] - mnew);
      mrow[r] = mnew;
      float rs = 0.0f;
#pragma unroll
      for (int n = 0; n < 8; ++n) {
        float p = __expf(s[n][r] - mnew);
        s[n][r] = p;
        rs += p;
      }
#pragma unroll
      for (int d = 1; d < 16; d <<= 1) rs += __shfl_xor(rs, d, 32);
      lrow[r] = lrow[r] * corr[r] + rs;
    }
#pragma unroll
    for (int d = 0; d < 4; ++d)
#pragma unroll
      for (int r = 0; r < 8; ++r) o[d][r] *= corr[r];

    __syncthreads();                    // all waves done reading sh.kt
    // Spill P (C layout) to LDS as bf16 so it can be re-read in A layout.
#pragma unroll
    for (int n = 0; n < 8; ++n)
#pragma unroll
      for (int r = 0; r < 8; ++r)
        sh.p[w][r + hi4 * 8][n * 16 + l15] = (__bf16)s[n][r];

    // O += P V : 4 k-steps over 128 keys x 4 d subtiles.
#pragma unroll
    for (int ks = 0; ks < 4; ++ks) {
      Frag pa;
      const bf16x8* prow = (const bf16x8*)&sh.p[w][l15][0];
      pa.h[0] = prow[ks * 4 + hi4];
      pa.h[1] = prow[ks * 4 + 2 + hi4];
#pragma unroll
      for (int ds = 0; ds < 4; ++ds) {
        Frag vb;
        const bf16x8* vrow = (const bf16x8*)&vt[ds * 16 + l15][0];
        vb.h[0] = vrow[ks * 4 + hi4];
        vb.h[1] = vrow[ks * 4 + 2 + hi4];
        o[ds] = WMMA_BF16(pa.v, vb.v, o[ds]);
      }
    }
  }

  // Normalize and store merged-head output [B,S,D] as bf16.
#pragma unroll
  for (int ds = 0; ds < 4; ++ds)
#pragma unroll
    for (int r = 0; r < 8; ++r) {
      int sq  = q0 + w * 16 + r + hi4 * 8;
      int col = h * kDK + ds * 16 + l15;
      float v = o[ds][r] / lrow[r];
      O[((size_t)b * kS + sq) * kD + col] = (__bf16)v;
    }
}

// ---------------------------------------------------------------------------
// Host-side orchestration.
// ---------------------------------------------------------------------------
extern "C" void kernel_launch(void* const* d_in, const int* in_sizes, int n_in,
                              void* d_out, int out_size, void* d_ws, size_t ws_size,
                              hipStream_t stream) {
  const float* query = (const float*)d_in[0];
  const float* key   = (const float*)d_in[1];
  const float* value = (const float*)d_in[2];
  // d_in[3] = mask (causal; applied analytically in-kernel)
  const float* Wq = (const float*)d_in[4];
  const float* bq = (const float*)d_in[5];
  const float* Wk = (const float*)d_in[6];
  const float* bk = (const float*)d_in[7];
  const float* Wv = (const float*)d_in[8];
  const float* bv = (const float*)d_in[9];
  const float* Wo = (const float*)d_in[10];
  const float* bo = (const float*)d_in[11];
  float* out = (float*)d_out;

  const int nAct = kM * kD;   // 8,388,608
  const int nW   = kD * kD;   // 1,048,576

  size_t off = 0;
  auto alloc = [&](size_t bytes) {
    void* p = (char*)d_ws + off;
    off += (bytes + 255) & ~(size_t)255;
    return p;
  };
  __bf16* xq   = (__bf16*)alloc((size_t)nAct * 2);
  __bf16* xk   = (__bf16*)alloc((size_t)nAct * 2);
  __bf16* xv   = (__bf16*)alloc((size_t)nAct * 2);
  __bf16* wqT  = (__bf16*)alloc((size_t)nW * 2);
  __bf16* wkT  = (__bf16*)alloc((size_t)nW * 2);
  __bf16* wvT  = (__bf16*)alloc((size_t)nW * 2);
  __bf16* woT  = (__bf16*)alloc((size_t)nW * 2);
  __bf16* Qws  = (__bf16*)alloc((size_t)nAct * 2);
  __bf16* Kws  = (__bf16*)alloc((size_t)nAct * 2);
  __bf16* Vws  = (__bf16*)alloc((size_t)nAct * 2);
  __bf16* attn = (__bf16*)alloc((size_t)nAct * 2);
  (void)ws_size; (void)n_in; (void)in_sizes; (void)out_size;

  // 1) fp32 -> bf16 conversions.
  cvt_bf16_kernel<<<nAct / 256, 256, 0, stream>>>(query, xq, nAct);
  cvt_bf16_kernel<<<nAct / 256, 256, 0, stream>>>(key,   xk, nAct);
  cvt_bf16_kernel<<<nAct / 256, 256, 0, stream>>>(value, xv, nAct);
  transpose_cvt_kernel<<<nW / 256, 256, 0, stream>>>(Wq, wqT);
  transpose_cvt_kernel<<<nW / 256, 256, 0, stream>>>(Wk, wkT);
  transpose_cvt_kernel<<<nW / 256, 256, 0, stream>>>(Wv, wvT);
  transpose_cvt_kernel<<<nW / 256, 256, 0, stream>>>(Wo, woT);

  // 2) Q/K/V projections (bf16 outputs).
  dim3 ggrid(kD / 128, kM / 128);
  gemm_bf16_kernel<<<ggrid, 256, 0, stream>>>(xq, wqT, bq, nullptr, Qws);
  gemm_bf16_kernel<<<ggrid, 256, 0, stream>>>(xk, wkT, bk, nullptr, Kws);
  gemm_bf16_kernel<<<ggrid, 256, 0, stream>>>(xv, wvT, bv, nullptr, Vws);

  // 3) Causal flash attention.
  dim3 agrid(kS / 128, kH, kB);
  attention_kernel<<<agrid, 256, 0, stream>>>(Qws, Kws, Vws, attn);

  // 4) Output projection (f32 + bias to d_out).
  gemm_bf16_kernel<<<ggrid, 256, 0, stream>>>(attn, woT, bo, out, nullptr);
}
